// MultiHeadAttention_14207751815746
// MI455X (gfx1250) — matrix operand order repaired and checked by reading, not compiled, not run on hardware
//
#include <hip/hip_runtime.h>
#include <hip/hip_bf16.h>
#include <math.h>

typedef __attribute__((ext_vector_type(16))) _Float16 v16h;
typedef __attribute__((ext_vector_type(8)))  _Float16 v8h;
typedef __attribute__((ext_vector_type(8)))  float    v8f;

#define BB    4
#define HH    256
#define WW    256
#define CC    64
#define NN    65536            // HH*WW
#define NHEAD 4
#define DHEAD 16
#define TSTR  40               // padded LDS row stride (halves) for [ch][token] tiles

static __device__ inline v8f wmma_f16(v16h a, v16h b, v8f c) {
  // v_wmma_f32_16x16x32_f16 : D = A(16x32 f16) x B(32x16 f16) + C(16x16 f32)
  return __builtin_amdgcn_wmma_f32_16x16x32_f16(
      /*neg_a=*/false, a, /*neg_b=*/false, b,
      /*c_mod=*/(short)0, c, /*reuse_a=*/false, /*reuse_b=*/false);
}

// Load the 8 loop-invariant B-tiles (32x16 KxN, f16) of a 64x64 row-major f32
// weight matrix straight from global into registers.
// B layout per ISA: VGPR r holds rows K = (r>>2)*16 + half*8 + 2*(r&3), +1 ; N = lane&15.
static __device__ inline void load_wtiles(const float* __restrict__ W, int l, int half,
                                          v16h out[2][4]) {
#pragma unroll
  for (int kc = 0; kc < 2; ++kc) {
#pragma unroll
    for (int nt = 0; nt < 4; ++nt) {
      v16h bb;
#pragma unroll
      for (int r = 0; r < 8; ++r) {
        const int k0 = kc * 32 + (r >> 2) * 16 + half * 8 + 2 * (r & 3);
        bb[2 * r]     = (_Float16)W[k0 * 64 + nt * 16 + l];
        bb[2 * r + 1] = (_Float16)W[(k0 + 1) * 64 + nt * 16 + l];
      }
      out[kc][nt] = bb;
    }
  }
}

// ---------------------------------------------------------------------------
// Kernel 0: zero the atomic accumulators (sumsq_q, sumsq_k, Gram)
// ---------------------------------------------------------------------------
__global__ void k0_init(float* ssq, float* ssk, float* G) {
  const int i = blockIdx.x * blockDim.x + threadIdx.x;
  if (i < BB * 64) { ssq[i] = 0.f; ssk[i] = 0.f; }
  if (i < BB * NHEAD * 16 * 16) G[i] = 0.f;
}

// ---------------------------------------------------------------------------
// Kernel 1: fused QKV projection (WMMA) + sum-of-squares + per-head Gram (WMMA)
//   - weights live in VGPRs (loop-invariant); q/k tiles bounce through LDS
//     in transposed [ch][token] layout with b128 vector access
//   - block = 128 threads = 4 waves, each wave owns 32-token tiles (grid-stride)
// ---------------------------------------------------------------------------
#define K1W 4
__global__ __launch_bounds__(128) void k1_qkv_gram(
    const float* __restrict__ x, const float* __restrict__ Wq,
    const float* __restrict__ Wk, const float* __restrict__ Wv,
    float* __restrict__ v_out, float* __restrict__ ssq,
    float* __restrict__ ssk, float* __restrict__ G) {
  __shared__ __align__(16) _Float16 sQ[K1W][64 * TSTR];  // [ch][token], padded
  __shared__ __align__(16) _Float16 sK[K1W][64 * TSTR];

  const int tid  = threadIdx.x;
  const int wave = tid >> 5;
  const int lane = tid & 31;
  const int l    = lane & 15;
  const int half = lane >> 4;
  const int b    = blockIdx.y;

  // loop-invariant weight tiles in registers (24 x v16h)
  v16h wQ[2][4], wK[2][4], wV[2][4];
  load_wtiles(Wq, l, half, wQ);
  load_wtiles(Wk, l, half, wK);
  load_wtiles(Wv, l, half, wV);

  v8f Gacc[NHEAD];
#pragma unroll
  for (int h = 0; h < NHEAD; ++h) Gacc[h] = {};
  float accq[4] = {0.f, 0.f, 0.f, 0.f};
  float acck[4] = {0.f, 0.f, 0.f, 0.f};

  const float* xb  = x + (size_t)b * NN * 64;
  float*       vb  = v_out + (size_t)b * NN * 64;
  _Float16*    myQ = sQ[wave];
  _Float16*    myK = sK[wave];

  const int tiles = NN / 32;
  for (int tile = blockIdx.x * K1W + wave; tile < tiles; tile += gridDim.x * K1W) {
    const int tok0 = tile * 32;

    // A tiles: x[token, ch] f32 -> f16, 2 token-halves x 2 K-chunks
    v16h a[2][2];
#pragma unroll
    for (int th = 0; th < 2; ++th) {
      const float* row = xb + (size_t)(tok0 + th * 16 + l) * 64;
#pragma unroll
      for (int kc = 0; kc < 2; ++kc) {
        v16h av;
#pragma unroll
        for (int r = 0; r < 8; ++r) {
          const int kb = kc * 32 + (r >> 2) * 16 + half * 8 + 2 * (r & 3);
          av[2 * r]     = (_Float16)row[kb];
          av[2 * r + 1] = (_Float16)row[kb + 1];
        }
        a[th][kc] = av;
      }
    }

    // projections: q,k -> LDS (+sumsq), v -> HBM
#pragma unroll
    for (int th = 0; th < 2; ++th) {
#pragma unroll
      for (int nt = 0; nt < 4; ++nt) {
        v8f cq = {};
        cq = wmma_f16(a[th][0], wQ[0][nt], cq);
        cq = wmma_f16(a[th][1], wQ[1][nt], cq);
        v8f ck = {};
        ck = wmma_f16(a[th][0], wK[0][nt], ck);
        ck = wmma_f16(a[th][1], wK[1][nt], ck);
        v8f cv = {};
        cv = wmma_f16(a[th][0], wV[0][nt], cv);
        cv = wmma_f16(a[th][1], wV[1][nt], cv);

        float sq = 0.f, sk = 0.f;
        v8h pq, pk;
#pragma unroll
        for (int r = 0; r < 8; ++r) {
          sq += cq[r] * cq[r];
          sk += ck[r] * ck[r];
          pq[r] = (_Float16)cq[r];
          pk[r] = (_Float16)ck[r];
          vb[(size_t)(tok0 + th * 16 + half * 8 + r) * 64 + nt * 16 + l] = cv[r];
        }
        accq[nt] += sq;
        acck[nt] += sk;
        // one b128 LDS store each: channel row (nt*16+l), tokens th*16+half*8 .. +7
        const int off = (nt * 16 + l) * TSTR + th * 16 + half * 8;
        *(v8h*)(myQ + off) = pq;
        *(v8h*)(myK + off) = pk;
      }
    }

    // per-head Gram: G[d,e] += k_tile^T (16d x 32tok) @ q_tile (32tok x 16e)
#pragma unroll
    for (int h = 0; h < NHEAD; ++h) {
      const _Float16* kRow = myK + (h * 16 + l) * TSTR;  // d = l
      const _Float16* qRow = myQ + (h * 16 + l) * TSTR;  // e = l
      const v8h alo = *(const v8h*)(kRow + half * 8);          // tokens half*8..+7
      const v8h ahi = *(const v8h*)(kRow + 16 + half * 8);     // tokens 16+half*8..+7
      const v8h blo = *(const v8h*)(qRow + half * 8);          // tokens half*8..+7
      const v8h bhi = *(const v8h*)(qRow + 16 + half * 8);     // tokens 16+half*8..+7
      v16h ga, gb;
#pragma unroll
      for (int i = 0; i < 8; ++i) {
        ga[i] = alo[i]; ga[8 + i] = ahi[i];
        gb[i] = blo[i]; gb[8 + i] = bhi[i];
      }
      Gacc[h] = wmma_f16(ga, gb, Gacc[h]);
    }
  }

  // flush reductions
#pragma unroll
  for (int nt = 0; nt < 4; ++nt) {
    atomicAdd(&ssq[b * 64 + nt * 16 + l], accq[nt]);
    atomicAdd(&ssk[b * 64 + nt * 16 + l], acck[nt]);
  }
#pragma unroll
  for (int h = 0; h < NHEAD; ++h) {
#pragma unroll
    for (int r = 0; r < 8; ++r) {
      atomicAdd(&G[((b * NHEAD + h) * 16 + half * 8 + r) * 16 + l], Gacc[h][r]);
    }
  }
}

// ---------------------------------------------------------------------------
// Kernel 2 (tiny, 1 block): normalize Gram -> softmax -> fold with W_proj
//   M[b][h*16+e][j] = sum_d attn[b,h,d,e] * W_proj[h*16+d][j]
// ---------------------------------------------------------------------------
__global__ __launch_bounds__(256) void k2_attn_M(
    const float* __restrict__ G, const float* __restrict__ ssq,
    const float* __restrict__ ssk, const float* __restrict__ rescale,
    const float* __restrict__ Wp, float* __restrict__ M) {
  __shared__ float sAttn[BB * NHEAD * 16 * 16];
  const int t = threadIdx.x;  // 256 threads exactly
  {
    const int b = t >> 6, h = (t >> 4) & 3, d = t & 15;
    const float Kd = fmaxf(sqrtf(ssk[b * 64 + h * 16 + d]), 1e-12f);
    const float rs = rescale[h];
    float row[16];
    float mx = -INFINITY;
#pragma unroll
    for (int e = 0; e < 16; ++e) {
      const float Qe = fmaxf(sqrtf(ssq[b * 64 + h * 16 + e]), 1e-12f);
      const float a  = G[((b * NHEAD + h) * 16 + d) * 16 + e] / (Kd * Qe) * rs;
      row[e] = a;
      mx = fmaxf(mx, a);
    }
    float s = 0.f;
#pragma unroll
    for (int e = 0; e < 16; ++e) { row[e] = __expf(row[e] - mx); s += row[e]; }
    const float inv = 1.f / s;
#pragma unroll
    for (int e = 0; e < 16; ++e)
      sAttn[((b * NHEAD + h) * 16 + d) * 16 + e] = row[e] * inv;
  }
  __syncthreads();
  {
    const int b = t >> 6, row = t & 63, h = row >> 4, e = row & 15;
    for (int j = 0; j < 64; ++j) {
      float m = 0.f;
#pragma unroll
      for (int d = 0; d < 16; ++d)
        m += sAttn[((b * NHEAD + h) * 16 + d) * 16 + e] * Wp[(h * 16 + d) * 64 + j];
      M[(b * 64 + row) * 64 + j] = m;
    }
  }
}

// ---------------------------------------------------------------------------
// Kernel 3: out_c = v_inp @ M_b + b_proj  (WMMA GEMM, M-tiles held in VGPRs)
// ---------------------------------------------------------------------------
__global__ __launch_bounds__(128) void k3_outproj(
    const float* __restrict__ v_inp, const float* __restrict__ M,
    const float* __restrict__ bproj, float* __restrict__ out) {
  const int tid  = threadIdx.x;
  const int wave = tid >> 5;
  const int lane = tid & 31;
  const int l    = lane & 15;
  const int half = lane >> 4;
  const int b    = blockIdx.y;

  v16h wM[2][4];
  load_wtiles(M + b * 4096, l, half, wM);
  float bias[4];
#pragma unroll
  for (int nt = 0; nt < 4; ++nt) bias[nt] = bproj[nt * 16 + l];

  const float* vbp = v_inp + (size_t)b * NN * 64;
  float*       ob  = out + (size_t)b * NN * 64;
  const int tiles = NN / 16;
  for (int tile = blockIdx.x * 4 + wave; tile < tiles; tile += gridDim.x * 4) {
    const int tok0 = tile * 16;
    const float* row = vbp + (size_t)(tok0 + l) * 64;
    v16h a0, a1;
#pragma unroll
    for (int r = 0; r < 8; ++r) {
      const int kb = (r >> 2) * 16 + half * 8 + 2 * (r & 3);
      a0[2 * r]     = (_Float16)row[kb];
      a0[2 * r + 1] = (_Float16)row[kb + 1];
      a1[2 * r]     = (_Float16)row[32 + kb];
      a1[2 * r + 1] = (_Float16)row[32 + kb + 1];
    }
#pragma unroll
    for (int nt = 0; nt < 4; ++nt) {
      v8f c = {};
      c = wmma_f16(a0, wM[0][nt], c);
      c = wmma_f16(a1, wM[1][nt], c);
#pragma unroll
      for (int r = 0; r < 8; ++r)
        ob[(size_t)(tok0 + half * 8 + r) * 64 + nt * 16 + l] = c[r] + bias[nt];
    }
  }
}

// ---------------------------------------------------------------------------
// Kernel 4a: p1 = gelu(dwconv3x3(v_inp, pe_w1))   (NHWC, lanes over channels)
// Kernel 4b: out += dwconv3x3(p1, pe_w2)
// ---------------------------------------------------------------------------
__global__ __launch_bounds__(256) void k4_conv_gelu(
    const float* __restrict__ in, const float* __restrict__ w,
    float* __restrict__ outp) {
  const int ch = threadIdx.x;                    // 0..63
  const int p  = blockIdx.x * 4 + threadIdx.y;   // pixel in [0, BB*NN)
  const int b  = p >> 16;
  const int rem = p & 65535;
  const int y = rem >> 8, xq = rem & 255;
  float acc = 0.f;
#pragma unroll
  for (int dy = 0; dy < 3; ++dy) {
#pragma unroll
    for (int dx = 0; dx < 3; ++dx) {
      const int yy = y + dy - 1, xx = xq + dx - 1;
      if (yy >= 0 && yy < HH && xx >= 0 && xx < WW)
        acc += in[((size_t)b * NN + yy * WW + xx) * 64 + ch] * w[ch * 9 + dy * 3 + dx];
    }
  }
  const float g = 0.5f * acc * (1.0f + erff(acc * 0.70710678118654752f));
  outp[(size_t)p * 64 + ch] = g;
}

__global__ __launch_bounds__(256) void k4_conv_add(
    const float* __restrict__ in, const float* __restrict__ w,
    float* __restrict__ out) {
  const int ch = threadIdx.x;
  const int p  = blockIdx.x * 4 + threadIdx.y;
  const int b  = p >> 16;
  const int rem = p & 65535;
  const int y = rem >> 8, xq = rem & 255;
  float acc = 0.f;
#pragma unroll
  for (int dy = 0; dy < 3; ++dy) {
#pragma unroll
    for (int dx = 0; dx < 3; ++dx) {
      const int yy = y + dy - 1, xx = xq + dx - 1;
      if (yy >= 0 && yy < HH && xx >= 0 && xx < WW)
        acc += in[((size_t)b * NN + yy * WW + xx) * 64 + ch] * w[ch * 9 + dy * 3 + dx];
    }
  }
  out[(size_t)p * 64 + ch] += acc;
}

// ---------------------------------------------------------------------------
extern "C" void kernel_launch(void* const* d_in, const int* in_sizes, int n_in,
                              void* d_out, int out_size, void* d_ws, size_t ws_size,
                              hipStream_t stream) {
  (void)in_sizes; (void)n_in; (void)out_size; (void)ws_size;
  const float* x    = (const float*)d_in[0];
  const float* Wq   = (const float*)d_in[1];
  const float* Wk   = (const float*)d_in[2];
  const float* Wv   = (const float*)d_in[3];
  const float* resc = (const float*)d_in[4];
  const float* Wp   = (const float*)d_in[5];
  const float* bp   = (const float*)d_in[6];
  const float* pw1  = (const float*)d_in[7];
  const float* pw2  = (const float*)d_in[8];
  float* out = (float*)d_out;

  float* ws    = (float*)d_ws;
  float* v_inp = ws;                                   // BB*NN*64
  float* p1    = v_inp + (size_t)BB * NN * 64;         // BB*NN*64
  float* ssq   = p1 + (size_t)BB * NN * 64;            // BB*64
  float* ssk   = ssq + BB * 64;                        // BB*64
  float* G     = ssk + BB * 64;                        // BB*NHEAD*256
  float* Mm    = G + BB * NHEAD * 256;                 // BB*64*64

  k0_init<<<dim3(16), dim3(256), 0, stream>>>(ssq, ssk, G);
  k1_qkv_gram<<<dim3(64, BB), dim3(128), 0, stream>>>(x, Wq, Wk, Wv, v_inp, ssq, ssk, G);
  k2_attn_M<<<dim3(1), dim3(256), 0, stream>>>(G, ssq, ssk, resc, Wp, Mm);
  k3_outproj<<<dim3(128, BB), dim3(128), 0, stream>>>(v_inp, Mm, bp, out);
  k4_conv_gelu<<<dim3((BB * NN) / 4), dim3(64, 4), 0, stream>>>(v_inp, pw1, p1);
  k4_conv_add<<<dim3((BB * NN) / 4), dim3(64, 4), 0, stream>>>(p1, pw2, out);
}
